// PredRNN_patch_32435593019602
// MI455X (gfx1250) — matrix operand order repaired and checked by reading, not compiled, not run on hardware
//
#include <hip/hip_runtime.h>

// ---------------------------------------------------------------------------
// CDNA5 (gfx1250) PredRNN forward: all convs lowered to implicit GEMM on
// v_wmma_f32_16x16x32_bf16 (wave32).  f32 recurrent state, bf16 conv operands.
// K ordering = (kh,kw) major / ci minor: each 32-K panel is one filter tap,
// so halo decode is uniform per panel and gathers are NSPAT-strided streams.
// ---------------------------------------------------------------------------

typedef __attribute__((ext_vector_type(16))) __bf16          v16bf;
typedef __attribute__((ext_vector_type(8)))  float           v8f;
typedef __attribute__((ext_vector_type(8)))  unsigned short  ushort8;

union BF16x16 { v16bf v; unsigned short u[16]; ushort8 h[2]; };

__device__ __forceinline__ unsigned short f2bf(float f) {
  unsigned int u = __builtin_bit_cast(unsigned int, f);
  u += 0x7FFFu + ((u >> 16) & 1u);           // round-to-nearest-even
  return (unsigned short)(u >> 16);
}
__device__ __forceinline__ float sigm(float x) { return 1.f / (1.f + __expf(-x)); }

#define NSPAT 8192      // B(8) * 32 * 32 pixels per frame
#define FRAME (64 * NSPAT)
#define MTILE 128
#define NTILE 64

// ---------------------------------------------------------------------------
// Implicit-GEMM conv: Y[M][8192] = W[M][K] * im2col(X[Cin][8192]) + bias
//   K = KS*KS*Cin with k = (kh*KS+kw)*Cin + ci.  SAME padding on 32x32 tiles.
//   Block: 256 threads = 8 waves; block tile 128(M) x 64(N); wave tile 16x64.
// ---------------------------------------------------------------------------
__global__ __launch_bounds__(256) void conv_wmma_bf16(
    const unsigned short* __restrict__ Wt,   // [M][K] bf16, tap-major K
    const unsigned short* __restrict__ Xt,   // [Cin][NSPAT] bf16 (+ z*inZ)
    const float* __restrict__ bias,          // [M]
    float* __restrict__ Yt,                  // [M][NSPAT] f32 (+ z*outZ)
    int M, int Cin, int KS, long inZ, long outZ)
{
  const int tid  = threadIdx.x;
  const int lane = tid & 31;
  const int wave = tid >> 5;
  const int hf   = (lane >> 4) & 1;   // lane half (wave32)
  const int l16  = lane & 15;
  const int K    = Cin * KS * KS;
  const unsigned short* X = Xt + (long)blockIdx.z * inZ;
  float* Y = Yt + (long)blockIdx.z * outZ;
  const int mBase = blockIdx.x * MTILE;
  const int nBase = blockIdx.y * NTILE;

  // stride 40 ushorts (80B): 16B-aligned chunks, lane rows tile all 64 banks
  __shared__ __attribute__((aligned(16))) unsigned short lA [MTILE][40]; // [m][k]
  __shared__ __attribute__((aligned(16))) unsigned short lBt[NTILE][40]; // [n][k]

  // ---- hoisted per-thread fill coordinates ----
  const int fn0 = tid & 63;            // B panel: owned column
  const int fg  = tid >> 6;            // B panel: ci-group of 8 (0..3)
  const int fn  = nBase + fn0;
  const int fb  = fn >> 10, fy = (fn >> 5) & 31, fx = fn & 31;
  const int fRow = tid >> 1;           // A panel: owned row (0..127)
  const int fc0  = tid & 1;            // A panel: chunk parity

  v8f acc[4] = {};

  for (int kb = 0; kb < K; kb += 32) {
    // uniform per-panel tap decode: panel lies inside one (kh,kw) segment
    const int r      = kb / Cin;             // kh*KS + kw (0 for KS==1)
    const int ciBase = kb - r * Cin + fg * 8;
    bool ok;
    long gbase;
    if (KS == 1) {
      ok = true;
      gbase = (long)ciBase * NSPAT + fn;
    } else {
      const int kh = r / 3, kw = r - kh * 3;
      const int iy = fy + kh - 1, ix = fx + kw - 1;
      ok = ((unsigned)iy < 32u) & ((unsigned)ix < 32u);
      gbase = (long)ciBase * NSPAT + (fb << 10) + (iy << 5) + ix;
    }

    __syncthreads();
    // ---- stage activation panel (column-major in LDS): strided gather ----
    {
      ushort8 vals = {0, 0, 0, 0, 0, 0, 0, 0};
      if (ok) {
#pragma unroll
        for (int i = 0; i < 8; ++i) vals[i] = X[gbase + (long)i * NSPAT];
      }
      *(ushort8*)&lBt[fn0][fg * 8] = vals;
    }
    // ---- stage weight panel: global_load_b128 -> ds_store_b128 ----
    {
      int row = mBase + fRow;
#pragma unroll
      for (int c = fc0; c < 4; c += 2) {
        ushort8 w = {0, 0, 0, 0, 0, 0, 0, 0};
        if (row < M) w = *(const ushort8*)&Wt[(long)row * K + kb + c * 8];
        *(ushort8*)&lA[fRow][c * 8] = w;
      }
    }
    // prefetch next K-panel of weights into cache
    if (kb + 32 < K) {
      int pr = mBase + fRow; if (pr >= M) pr = M - 1;
      __builtin_prefetch(Wt + (long)pr * K + kb + 32 + fc0 * 16, 0, 3);
    }
    __syncthreads();

    // ---- A fragment: rows lane%16, K chunks {hf*8..+7} and {16+hf*8..+7} ----
    BF16x16 afr;
    afr.h[0] = *(const ushort8*)&lA[wave * 16 + l16][hf * 8];
    afr.h[1] = *(const ushort8*)&lA[wave * 16 + l16][16 + hf * 8];

    // ---- 4 N-subtiles: B fragment = K chunks {hf*8..+7} and {16+hf*8..+7} ----
#pragma unroll
    for (int j = 0; j < 4; ++j) {
      BF16x16 bfr;
      bfr.h[0] = *(const ushort8*)&lBt[j * 16 + l16][hf * 8];
      bfr.h[1] = *(const ushort8*)&lBt[j * 16 + l16][16 + hf * 8];
      acc[j] = __builtin_amdgcn_wmma_f32_16x16x32_bf16(
          false, afr.v, false, bfr.v, (short)0, acc[j], false, false);
    }
  }

  // ---- epilogue: bias add + store (C layout: row = r + 8*half, col = lane%16)
  const int mRow0 = mBase + wave * 16;
#pragma unroll
  for (int j = 0; j < 4; ++j) {
    int n = nBase + j * 16 + l16;
#pragma unroll
    for (int r = 0; r < 8; ++r) {
      int row = mRow0 + r + 8 * hf;
      if (row < M) Y[(long)row * NSPAT + n] = acc[j][r] + bias[row];
    }
  }
}

// ---------------------------------------------------------------------------
// Weight convert + permute: src [M][Cin][KS][KS] f32 -> dst [M][K] bf16,
// K ordered tap-major: k = (kh*KS+kw)*Cin + ci  (identity for KS==1)
// ---------------------------------------------------------------------------
__global__ void wperm_k(const float* __restrict__ src, unsigned short* __restrict__ dst,
                        int M, int Cin, int KS) {
  const int K = Cin * KS * KS;
  long total = (long)M * K;
  long i = (long)blockIdx.x * 256 + threadIdx.x;
  if (i >= total) return;
  long m = i / K;
  int  k = (int)(i - m * K);
  int  r = k / Cin, ci = k - r * Cin;
  dst[i] = f2bf(src[(m * Cin + ci) * (KS * KS) + r]);
}

// x(8,10,1,256,256) -> xp bf16 [t][ch=a*8+b8][b][32][32]
__global__ void patchify_k(const float* __restrict__ x, unsigned short* __restrict__ xp) {
  long i = (long)blockIdx.x * 256 + threadIdx.x;
  if (i >= 10L * FRAME) return;
  int n = i & (NSPAT - 1), ch = (int)((i >> 13) & 63), t = (int)(i >> 19);
  int b = n >> 10, ii = (n >> 5) & 31, jj = n & 31;
  int a = ch >> 3, b8 = ch & 7;
  long src = ((long)(b * 10 + t) * 256 + (a * 32 + ii)) * 256 + (b8 * 32 + jj);
  xp[i] = f2bf(x[src]);
}

// gates part 1: i/f/g + i'/f'/g' -> c_new, m_new, o_pre, bf16 mem & m
__global__ void gates1_k(const float* __restrict__ xc, const float* __restrict__ hc,
                         const float* __restrict__ mc, float* __restrict__ c,
                         float* __restrict__ m, float* __restrict__ o_pre,
                         unsigned short* __restrict__ mem_bf,
                         unsigned short* __restrict__ m_bf) {
  int i = blockIdx.x * 256 + threadIdx.x;
  if (i >= FRAME) return;
  int ch = i >> 13, n = i & (NSPAT - 1);
  const int S = NSPAT;
  float ix  = xc[(0 * 64 + ch) * S + n], fx  = xc[(1 * 64 + ch) * S + n];
  float gx  = xc[(2 * 64 + ch) * S + n], ixp = xc[(3 * 64 + ch) * S + n];
  float fxp = xc[(4 * 64 + ch) * S + n], gxp = xc[(5 * 64 + ch) * S + n];
  float ox  = xc[(6 * 64 + ch) * S + n];
  float ih = hc[ch * S + n],        fh = hc[(64 + ch) * S + n];
  float gh = hc[(128 + ch) * S + n], oh = hc[(192 + ch) * S + n];
  float im = mc[ch * S + n], fm = mc[(64 + ch) * S + n], gm = mc[(128 + ch) * S + n];
  float cn = sigm(fx + fh) * c[i] + sigm(ix + ih) * tanhf(gx + gh);
  float mn = sigm(fxp + fm) * m[i] + sigm(ixp + im) * tanhf(gxp + gm);
  c[i] = cn; m[i] = mn; o_pre[i] = ox + oh;
  mem_bf[i] = f2bf(cn); mem_bf[FRAME + i] = f2bf(mn);
  m_bf[i] = f2bf(mn);
}

// gates part 2: o = sig(o_pre + oc); h = o * tanh(lc)
__global__ void gates2_k(const float* __restrict__ o_pre, const float* __restrict__ oc,
                         const float* __restrict__ lc, unsigned short* __restrict__ h_bf) {
  int i = blockIdx.x * 256 + threadIdx.x;
  if (i >= FRAME) return;
  float h = sigm(o_pre[i] + oc[i]) * tanhf(lc[i]);
  h_bf[i] = f2bf(h);
}

// decoder: sigmoid + depth-to-space 8x8 into (B,19,1,256,256) f32
__global__ void dec_scatter_k(const float* __restrict__ yc, float* __restrict__ out) {
  long i = (long)blockIdx.x * 256 + threadIdx.x;
  if (i >= 19L * FRAME) return;
  int n = (int)(i & (NSPAT - 1)), ch = (int)((i >> 13) & 63), t = (int)(i >> 19);
  int b = n >> 10, ii = (n >> 5) & 31, jj = n & 31;
  int a = ch >> 3, b8 = ch & 7;
  long dst = ((long)(b * 19 + t) * 256 + (a * 32 + ii)) * 256 + (b8 * 32 + jj);
  out[dst] = sigm(yc[i]);
}

// ---------------------------------------------------------------------------
// Host orchestration
// ---------------------------------------------------------------------------
extern "C" void kernel_launch(void* const* d_in, const int* in_sizes, int n_in,
                              void* d_out, int out_size, void* d_ws, size_t ws_size,
                              hipStream_t stream) {
  (void)in_sizes; (void)n_in; (void)out_size; (void)ws_size;
  const float* x  = (const float*)d_in[0];
  const float* Wx = (const float*)d_in[3];  const float* bx = (const float*)d_in[4];
  const float* Wh = (const float*)d_in[5];  const float* bh = (const float*)d_in[6];
  const float* Wm = (const float*)d_in[7];  const float* bm = (const float*)d_in[8];
  const float* Wo = (const float*)d_in[9];  const float* bo = (const float*)d_in[10];
  const float* Wl = (const float*)d_in[11]; const float* bl = (const float*)d_in[12];
  const float* Wd = (const float*)d_in[13]; const float* bd = (const float*)d_in[14];
  float* out = (float*)d_out;

  char* p = (char*)d_ws;
  auto carve = [&](size_t bytes) -> char* {
    char* r = p; p += (bytes + 255) & ~(size_t)255; return r;
  };
  const long KWX = 576, KWO = 1152, KWL = 128;
  unsigned short* Wxb = (unsigned short*)carve(4L * 448 * KWX * 2);
  unsigned short* Whb = (unsigned short*)carve(4L * 256 * KWX * 2);
  unsigned short* Wmb = (unsigned short*)carve(4L * 192 * KWX * 2);
  unsigned short* Wob = (unsigned short*)carve(4L * 64 * KWO * 2);
  unsigned short* Wlb = (unsigned short*)carve(4L * 64 * KWL * 2);
  unsigned short* Wdb = (unsigned short*)carve(64L * KWX * 2);
  unsigned short* xp   = (unsigned short*)carve(10L * FRAME * 2);
  unsigned short* outs = (unsigned short*)carve(19L * FRAME * 2);
  unsigned short* hbf  = (unsigned short*)carve(3L * FRAME * 2);  // layers 0..2
  unsigned short* h3z  = (unsigned short*)carve((long)FRAME * 2); // zero init h[3]
  unsigned short* m_bf = (unsigned short*)carve((long)FRAME * 2);
  unsigned short* membf= (unsigned short*)carve(2L * FRAME * 2);
  float* cbuf  = (float*)carve(4L * FRAME * 4);
  float* mbuf  = (float*)carve((long)FRAME * 4);
  float* xcb   = (float*)carve(7L * FRAME * 4);
  float* hcb   = (float*)carve(4L * FRAME * 4);
  float* mcb   = (float*)carve(3L * FRAME * 4);
  float* opre  = (float*)carve((long)FRAME * 4);
  float* ocb   = (float*)carve((long)FRAME * 4);
  float* lcb   = (float*)carve((long)FRAME * 4);
  float* ycb   = (float*)carve(19L * FRAME * 4);

  // zero-init recurrent state (ws is poisoned by the harness)
  hipMemsetAsync(cbuf, 0, 4L * FRAME * 4, stream);
  hipMemsetAsync(mbuf, 0, (long)FRAME * 4, stream);
  hipMemsetAsync(m_bf, 0, (long)FRAME * 2, stream);
  hipMemsetAsync(hbf,  0, 3L * FRAME * 2, stream);
  hipMemsetAsync(h3z,  0, (long)FRAME * 2, stream);

  auto wperm = [&](const float* src, unsigned short* dst, int M, int Cin, int KS) {
    long total = (long)M * Cin * KS * KS;
    wperm_k<<<dim3((unsigned)((total + 255) / 256)), dim3(256), 0, stream>>>(
        src, dst, M, Cin, KS);
  };
  wperm(Wx, Wxb, 4 * 448, 64, 3);
  wperm(Wh, Whb, 4 * 256, 64, 3);
  wperm(Wm, Wmb, 4 * 192, 64, 3);
  wperm(Wo, Wob, 4 * 64, 128, 3);
  wperm(Wl, Wlb, 4 * 64, 128, 1);
  wperm(Wd, Wdb, 64, 64, 3);
  patchify_k<<<dim3((unsigned)((10L * FRAME + 255) / 256)), dim3(256), 0, stream>>>(x, xp);

  auto conv = [&](const unsigned short* Wt, const unsigned short* Xt, const float* bias,
                  float* Yt, int M, int Cin, int KS, long inZ, long outZ, int Z) {
    dim3 g((M + MTILE - 1) / MTILE, NSPAT / NTILE, Z);
    conv_wmma_bf16<<<g, dim3(256), 0, stream>>>(Wt, Xt, bias, Yt, M, Cin, KS, inZ, outZ);
  };
  const dim3 GE((FRAME + 255) / 256), TB(256);

  for (int t = 0; t < 19; ++t) {
    const unsigned short* cur = (t < 10) ? xp + (long)t * FRAME
                                         : outs + (long)(t - 1) * FRAME;
    for (int l = 0; l < 4; ++l) {
      const unsigned short* hprev =
          (l == 3) ? ((t == 0) ? h3z : outs + (long)(t - 1) * FRAME)
                   : hbf + (long)l * FRAME;
      conv(Wxb + (long)l * 448 * KWX, cur,   bx + l * 448, xcb, 448, 64, 3, 0, 0, 1);
      conv(Whb + (long)l * 256 * KWX, hprev, bh + l * 256, hcb, 256, 64, 3, 0, 0, 1);
      conv(Wmb + (long)l * 192 * KWX, m_bf,  bm + l * 192, mcb, 192, 64, 3, 0, 0, 1);
      gates1_k<<<GE, TB, 0, stream>>>(xcb, hcb, mcb, cbuf + (long)l * FRAME, mbuf,
                                      opre, membf, m_bf);
      conv(Wob + (long)l * 64 * KWO, membf, bo + l * 64, ocb, 64, 128, 3, 0, 0, 1);
      conv(Wlb + (long)l * 64 * KWL, membf, bl + l * 64, lcb, 64, 128, 1, 0, 0, 1);
      unsigned short* hout = (l == 3) ? outs + (long)t * FRAME : hbf + (long)l * FRAME;
      gates2_k<<<GE, TB, 0, stream>>>(opre, ocb, lcb, hout);
      cur = hout;
    }
  }

  // decoder conv over all 19 frames in one launch (gridDim.z), then scatter
  conv(Wdb, outs, bd, ycb, 64, 64, 3, (long)FRAME, (long)FRAME, 19);
  dec_scatter_k<<<dim3((unsigned)((19L * FRAME + 255) / 256)), dim3(256), 0, stream>>>(ycb, out);
}
